// GPTBigCodeBlock_53249004536102
// MI455X (gfx1250) — matrix operand order repaired and checked by reading, not compiled, not run on hardware
//
#include <hip/hip_runtime.h>
#include <hip/hip_bf16.h>
#include <math.h>
#include <stdint.h>

typedef __bf16 bf16;
typedef __attribute__((ext_vector_type(16))) __bf16 v16bf;
typedef __attribute__((ext_vector_type(8)))  __bf16 v8bf;
typedef __attribute__((ext_vector_type(8)))  float  v8f;
typedef __attribute__((ext_vector_type(4)))  int    v4i;

#define HID    2048
#define NHEAD  16
#define HD     128
#define INTERN 8192
#define SEQ    2048
#define NBATCH 4
#define NTOK   (NBATCH * SEQ)
#define LN_EPS 1e-5f

// ---- CDNA5 async global->LDS staging (guarded; falls back to VGPR copy) ----
#if defined(__has_builtin)
#if __has_builtin(__builtin_amdgcn_global_load_async_to_lds_b128) && \
    __has_builtin(__builtin_amdgcn_s_wait_asynccnt)
#define HAVE_ASYNC_LDS 1
#endif
#endif
#ifndef HAVE_ASYNC_LDS
#define HAVE_ASYNC_LDS 0
#endif

#define AS1 __attribute__((address_space(1)))
#define AS3 __attribute__((address_space(3)))

// copy 16 bytes global -> LDS (per active lane)
static __device__ inline void g2lds_b128(const bf16* g, bf16* l) {
#if HAVE_ASYNC_LDS
    // low 32 bits of a generic LDS pointer are the LDS byte offset (ISA 10.2)
    __builtin_amdgcn_global_load_async_to_lds_b128(
        (AS1 v4i*)(uintptr_t)g, (AS3 v4i*)(uint32_t)(uintptr_t)l, 0, 0);
#else
    *(int4*)l = *(const int4*)g;
#endif
}
static __device__ inline void async_stage_wait() {
#if HAVE_ASYNC_LDS
    __builtin_amdgcn_s_wait_asynccnt(0);
#endif
}

static __device__ inline v8f v8f_zero() {
    v8f r;
#pragma unroll
    for (int i = 0; i < 8; i++) r[i] = 0.0f;
    return r;
}

static __device__ inline v16bf cat8(v8bf lo, v8bf hi) {
    v16bf r;
#pragma unroll
    for (int i = 0; i < 8; i++) { r[i] = lo[i]; r[i + 8] = hi[i]; }
    return r;
}

static __device__ inline float gelu_tanh(float x) {
    float x3 = x * x * x;
    return 0.5f * x * (1.0f + tanhf(0.79788456080286535588f * (x + 0.044715f * x3)));
}

// ---------------------------------------------------------------------------
// fp32 -> bf16 conversion (grid-stride)
// ---------------------------------------------------------------------------
__global__ void cvt_f32_bf16_kernel(const float* __restrict__ in,
                                    bf16* __restrict__ out, long n) {
    long i  = (long)blockIdx.x * blockDim.x + threadIdx.x;
    long st = (long)gridDim.x * blockDim.x;
    for (; i < n; i += st) out[i] = (bf16)in[i];
}

// ---------------------------------------------------------------------------
// LayerNorm over H=2048, one 256-thread block per token, bf16 output
// ---------------------------------------------------------------------------
__global__ __launch_bounds__(256)
void layernorm_bf16_kernel(const float* __restrict__ x, const float* __restrict__ g,
                           const float* __restrict__ bta, bf16* __restrict__ out) {
    __shared__ float ws[8], ws2[8];
    const int  t   = threadIdx.x;
    const long tok = blockIdx.x;
    const float* xr = x + tok * HID;

    float s = 0.f, s2 = 0.f;
    float v[8];
#pragma unroll
    for (int i = 0; i < 8; i++) {
        v[i] = xr[t + i * 256];
        s += v[i]; s2 += v[i] * v[i];
    }
#pragma unroll
    for (int off = 1; off < 32; off <<= 1) {
        s  += __shfl_xor(s,  off, 32);
        s2 += __shfl_xor(s2, off, 32);
    }
    if ((t & 31) == 0) { ws[t >> 5] = s; ws2[t >> 5] = s2; }
    __syncthreads();
    s = 0.f; s2 = 0.f;
#pragma unroll
    for (int j = 0; j < 8; j++) { s += ws[j]; s2 += ws2[j]; }
    const float mean = s  * (1.0f / HID);
    const float var  = s2 * (1.0f / HID) - mean * mean;
    const float rstd = rsqrtf(var + LN_EPS);

    bf16* orow = out + tok * HID;
#pragma unroll
    for (int i = 0; i < 8; i++) {
        int c = t + i * 256;
        orow[c] = (bf16)((v[i] - mean) * rstd * g[c] + bta[c]);
    }
}

// ---------------------------------------------------------------------------
// bf16 WMMA GEMM: C[M,N] = A[M,K] @ B[K,N] + bias (+gelu) (+fp32 residual)
// Block tile 128x128, 256 threads = 8 waves (4 along M x 2 along N),
// wave tile 32x64 = 2x4 WMMA tiles -> 8 WMMAs per 12 fragment ds_load_b128.
// A staged [128][32] row-major (async global->LDS when available);
// B staged transposed [128][32] so fragments are contiguous 16B LDS reads.
// ---------------------------------------------------------------------------
__global__ __launch_bounds__(256)
void gemm_bf16_kernel(const bf16* __restrict__ A, const bf16* __restrict__ Bw,
                      const float* __restrict__ bias, const float* __restrict__ resid,
                      float* __restrict__ outF, bf16* __restrict__ outB,
                      int M, int N, int K, int act_gelu) {
    __shared__ __align__(16) bf16 As[128][32];
    __shared__ __align__(16) bf16 Bs[128][32];   // [N][K] transposed

    const int t    = threadIdx.x;
    const int lane = t & 31;
    const int w    = t >> 5;
    const int wm   = w & 3;           // 0..3 along M (32 rows each)
    const int wn   = w >> 2;          // 0..1 along N (64 cols each)
    const int blockN = blockIdx.x * 128;
    const int blockM = blockIdx.y * 128;

    v8f acc[2][4];
#pragma unroll
    for (int i = 0; i < 2; i++)
#pragma unroll
        for (int j = 0; j < 4; j++) acc[i][j] = v8f_zero();

    const int arow  = t >> 1;          // 0..127
    const int ahalf = (t & 1) * 16;    // 0 or 16
    const int brow  = t >> 3;          // 0..31  (K row)
    const int bn0   = (t & 7) * 16;    // 0..112 (N cols, 16 each)

    const int fr  = lane & 15;
    const int akb = (lane < 16) ? 0 : 16;  // A-fragment K base (elements)
    const int bkb = (lane < 16) ? 0 : 16;  // B-fragment K base (elements)

    for (int k0 = 0; k0 < K; k0 += 32) {
        __syncthreads();
        // --- stage A tile (async global->LDS on CDNA5) ---
        const bf16* ag = A + (size_t)(blockM + arow) * K + k0 + ahalf;
        g2lds_b128(ag,     &As[arow][ahalf]);
        g2lds_b128(ag + 8, &As[arow][ahalf + 8]);
        // --- stage B tile, transposed into [N][K] ---
        const bf16* bg = Bw + (size_t)(k0 + brow) * N + blockN + bn0;
        bf16 tb[16];
        *(int4*)&tb[0] = *(const int4*)bg;
        *(int4*)&tb[8] = *(const int4*)(bg + 8);
#pragma unroll
        for (int j = 0; j < 16; j++) Bs[bn0 + j][brow] = tb[j];
        if (k0 + 32 < K) __builtin_prefetch(ag + 32, 0, 1);
        async_stage_wait();
        __syncthreads();

        v16bf af[2], bfrag[4];
#pragma unroll
        for (int mt = 0; mt < 2; mt++) {
            const bf16* ap = &As[wm * 32 + mt * 16 + fr][0];
            af[mt] = cat8(*(const v8bf*)(ap + akb), *(const v8bf*)(ap + akb + 8));
        }
#pragma unroll
        for (int nt = 0; nt < 4; nt++) {
            const bf16* bp = &Bs[wn * 64 + nt * 16 + fr][0];
            bfrag[nt] = cat8(*(const v8bf*)(bp + bkb), *(const v8bf*)(bp + bkb + 8));
        }
#pragma unroll
        for (int mt = 0; mt < 2; mt++)
#pragma unroll
            for (int nt = 0; nt < 4; nt++)
                acc[mt][nt] = __builtin_amdgcn_wmma_f32_16x16x32_bf16(
                    false, af[mt], false, bfrag[nt], (short)0, acc[mt][nt], false, false);
    }

    // --- epilogue: bias (+gelu) (+residual), fp32 and/or bf16 output ---
#pragma unroll
    for (int mt = 0; mt < 2; mt++) {
#pragma unroll
        for (int nt = 0; nt < 4; nt++) {
#pragma unroll
            for (int r = 0; r < 8; r++) {
                const int row = blockM + wm * 32 + mt * 16 + ((lane < 16) ? r : 8 + r);
                const int col = blockN + wn * 64 + nt * 16 + fr;
                float vv = acc[mt][nt][r] + bias[col];
                if (act_gelu) vv = gelu_tanh(vv);
                const size_t idx = (size_t)row * N + col;
                if (resid) vv += resid[idx];
                if (outF) outF[idx] = vv;
                if (outB) outB[idx] = (bf16)vv;
            }
        }
    }
}

// ---------------------------------------------------------------------------
// Causal multi-query flash attention.
// Grid (S/64, NHEAD, NBATCH), 128 threads = 4 waves; wave owns 16 query rows.
// 32-key steps: K staged via async global->LDS, V staged transposed,
// QK^T via 8 chained WMMAs, online softmax with half-wave shuffles,
// P->LDS->A-fragment, P@V 8 WMMAs.
// ---------------------------------------------------------------------------
__global__ __launch_bounds__(128)
void attn_kernel(const bf16* __restrict__ qb, const bf16* __restrict__ kvb,
                 bf16* __restrict__ ob) {
    __shared__ __align__(16) bf16 Ks[32][128];
    __shared__ __align__(16) bf16 Vs[128][32];
    __shared__ __align__(16) bf16 Ps[4][16][32];

    const int t    = threadIdx.x;
    const int lane = t & 31;
    const int w    = t >> 5;          // 0..3
    const int qt   = blockIdx.x;      // 0..31
    const int h    = blockIdx.y;      // 0..15
    const int b    = blockIdx.z;      // 0..3

    const int  q0    = qt * 64 + w * 16;           // seq pos of wave's first q row
    const long tokq0 = (long)b * SEQ + q0;
    const int  qlast = q0 + 15;
    const float scale = 0.08838834764831845f;      // 1/sqrt(128)

    const int fr  = lane & 15;
    const int akb = (lane < 16) ? 0 : 16;
    const int bkb = (lane < 16) ? 0 : 16;

    // Q fragments (A layout), head_dim 128 = 4 chunks of K=32
    v16bf qf[4];
#pragma unroll
    for (int dc = 0; dc < 4; dc++) {
        const bf16* qg = qb + (size_t)(tokq0 + fr) * HID + h * HD + dc * 32;
        qf[dc] = cat8(*(const v8bf*)(qg + akb), *(const v8bf*)(qg + akb + 8));
    }

    v8f acc[8];
#pragma unroll
    for (int ct = 0; ct < 8; ct++) acc[ct] = v8f_zero();
    float mrow[8], lrow[8];
#pragma unroll
    for (int r = 0; r < 8; r++) { mrow[r] = -1e30f; lrow[r] = 0.0f; }

    const int kend = qt * 64 + 64;
    const int skey = t >> 2;           // 0..31
    const int sdb  = (t & 3) * 32;     // 0,32,64,96

    for (int ks = 0; ks < kend; ks += 32) {
        __syncthreads();
        {   // stage K [32][128] (async) and V transposed [128][32]
            const bf16* kg = kvb + (size_t)((long)b * SEQ + ks + skey) * 256 + sdb;
#pragma unroll
            for (int i = 0; i < 4; i++)
                g2lds_b128(kg + i * 8, &Ks[skey][sdb + i * 8]);
            const bf16* vg = kg + 128;
            bf16 tv[32];
#pragma unroll
            for (int i = 0; i < 4; i++)
                *(int4*)&tv[i * 8] = *(const int4*)(vg + i * 8);
#pragma unroll
            for (int j = 0; j < 32; j++) Vs[sdb + j][skey] = tv[j];
        }
        async_stage_wait();
        __syncthreads();

        if (ks <= qlast) {
            // scores for two 16-key tiles
            v8f s0 = v8f_zero(), s1 = v8f_zero();
#pragma unroll
            for (int dc = 0; dc < 4; dc++) {
                const bf16* kp0 = &Ks[fr][dc * 32 + bkb];
                v16bf kf0 = cat8(*(const v8bf*)kp0, *(const v8bf*)(kp0 + 8));
                s0 = __builtin_amdgcn_wmma_f32_16x16x32_bf16(
                    false, qf[dc], false, kf0, (short)0, s0, false, false);
                const bf16* kp1 = &Ks[16 + fr][dc * 32 + bkb];
                v16bf kf1 = cat8(*(const v8bf*)kp1, *(const v8bf*)(kp1 + 8));
                s1 = __builtin_amdgcn_wmma_f32_16x16x32_bf16(
                    false, qf[dc], false, kf1, (short)0, s1, false, false);
            }
            // online softmax per query row (rows live in VGPR index, cols in lanes)
#pragma unroll
            for (int r = 0; r < 8; r++) {
                const int rr   = (lane < 16) ? r : 8 + r;
                const int qpos = q0 + rr;
                float a0 = s0[r] * scale; if (ks + fr      > qpos) a0 = -1e30f;
                float a1 = s1[r] * scale; if (ks + 16 + fr > qpos) a1 = -1e30f;
                float mr = fmaxf(a0, a1);
#pragma unroll
                for (int off = 1; off < 16; off <<= 1)
                    mr = fmaxf(mr, __shfl_xor(mr, off, 16));
                const float nm    = fmaxf(mrow[r], mr);
                const float alpha = __expf(mrow[r] - nm);
                const float p0 = __expf(a0 - nm);
                const float p1 = __expf(a1 - nm);
                float rs = p0 + p1;
#pragma unroll
                for (int off = 1; off < 16; off <<= 1)
                    rs += __shfl_xor(rs, off, 16);
                lrow[r] = lrow[r] * alpha + rs;
                mrow[r] = nm;
#pragma unroll
                for (int ct = 0; ct < 8; ct++) acc[ct][r] *= alpha;
                Ps[w][rr][fr]      = (bf16)p0;
                Ps[w][rr][16 + fr] = (bf16)p1;
            }
            // P (A layout) @ V
            const bf16* pp = &Ps[w][fr][0];
            v16bf pf = cat8(*(const v8bf*)(pp + akb), *(const v8bf*)(pp + akb + 8));
#pragma unroll
            for (int ct = 0; ct < 8; ct++) {
                const bf16* vp = &Vs[ct * 16 + fr][bkb];
                v16bf vf = cat8(*(const v8bf*)vp, *(const v8bf*)(vp + 8));
                acc[ct] = __builtin_amdgcn_wmma_f32_16x16x32_bf16(
                    false, pf, false, vf, (short)0, acc[ct], false, false);
            }
        }
    }

    // normalize and write bf16 output [token][HID] at this head's columns
#pragma unroll
    for (int ct = 0; ct < 8; ct++) {
#pragma unroll
        for (int r = 0; r < 8; r++) {
            const int rr = (lane < 16) ? r : 8 + r;
            float o = acc[ct][r] / lrow[r];
            ob[(size_t)(tokq0 + rr) * HID + h * HD + ct * 16 + fr] = (bf16)o;
        }
    }
}

// ---------------------------------------------------------------------------
// Orchestration
// ---------------------------------------------------------------------------
extern "C" void kernel_launch(void* const* d_in, const int* in_sizes, int n_in,
                              void* d_out, int out_size, void* d_ws, size_t ws_size,
                              hipStream_t stream) {
    (void)in_sizes; (void)n_in; (void)out_size; (void)ws_size;
    const float* x     = (const float*)d_in[0];
    const float* ln1_g = (const float*)d_in[1];
    const float* ln1_b = (const float*)d_in[2];
    const float* ln2_g = (const float*)d_in[3];
    const float* ln2_b = (const float*)d_in[4];
    const float* wq    = (const float*)d_in[5];
    const float* bq    = (const float*)d_in[6];
    const float* wkv   = (const float*)d_in[7];
    const float* bkv   = (const float*)d_in[8];
    const float* wo    = (const float*)d_in[9];
    const float* bo    = (const float*)d_in[10];
    const float* wfc   = (const float*)d_in[11];
    const float* bfc   = (const float*)d_in[12];
    const float* wproj = (const float*)d_in[13];
    const float* bproj = (const float*)d_in[14];

    char* ws = (char*)d_ws;
    size_t off = 0;
    auto take = [&](size_t bytes) -> char* {
        char* p = ws + off;
        off += (bytes + 255) & ~(size_t)255;
        return p;
    };

    bf16*  wq_b    = (bf16*) take((size_t)HID * HID * 2);
    bf16*  wkv_b   = (bf16*) take((size_t)HID * 256 * 2);
    bf16*  wo_b    = (bf16*) take((size_t)HID * HID * 2);
    bf16*  wfc_b   = (bf16*) take((size_t)HID * INTERN * 2);
    bf16*  wproj_b = (bf16*) take((size_t)INTERN * HID * 2);
    bf16*  hln     = (bf16*) take((size_t)NTOK * HID * 2);
    bf16*  qbuf    = (bf16*) take((size_t)NTOK * HID * 2);
    bf16*  kvbuf   = (bf16*) take((size_t)NTOK * 256 * 2);
    bf16*  attnb   = (bf16*) take((size_t)NTOK * HID * 2);
    float* x1      = (float*)take((size_t)NTOK * HID * 4);
    bf16*  fcb     = (bf16*) take((size_t)NTOK * INTERN * 2);

    auto cvt = [&](const float* src, bf16* dst, long n) {
        long grid = (n + 256 * 8 - 1) / (256 * 8);
        if (grid > 16384) grid = 16384;
        cvt_f32_bf16_kernel<<<(int)grid, 256, 0, stream>>>(src, dst, n);
    };
    cvt(wq,    wq_b,    (long)HID * HID);
    cvt(wkv,   wkv_b,   (long)HID * 256);
    cvt(wo,    wo_b,    (long)HID * HID);
    cvt(wfc,   wfc_b,   (long)HID * INTERN);
    cvt(wproj, wproj_b, (long)INTERN * HID);

    // LN1 -> bf16
    layernorm_bf16_kernel<<<NTOK, 256, 0, stream>>>(x, ln1_g, ln1_b, hln);

    // Q = hln @ wq + bq  (bf16 out)
    gemm_bf16_kernel<<<dim3(HID / 128, NTOK / 128), 256, 0, stream>>>(
        hln, wq_b, bq, nullptr, nullptr, qbuf, NTOK, HID, HID, 0);
    // KV = hln @ wkv + bkv (bf16 out)
    gemm_bf16_kernel<<<dim3(256 / 128, NTOK / 128), 256, 0, stream>>>(
        hln, wkv_b, bkv, nullptr, nullptr, kvbuf, NTOK, 256, HID, 0);

    // causal MQA flash attention -> attnb (bf16)
    attn_kernel<<<dim3(SEQ / 64, NHEAD, NBATCH), 128, 0, stream>>>(qbuf, kvbuf, attnb);

    // x1 = attnb @ wo + bo + x (fp32 residual stream)
    gemm_bf16_kernel<<<dim3(HID / 128, NTOK / 128), 256, 0, stream>>>(
        attnb, wo_b, bo, x, x1, nullptr, NTOK, HID, HID, 0);

    // LN2 -> bf16 (reuse hln)
    layernorm_bf16_kernel<<<NTOK, 256, 0, stream>>>(x1, ln2_g, ln2_b, hln);

    // fcb = gelu(hln @ wfc + bfc) (bf16)
    gemm_bf16_kernel<<<dim3(INTERN / 128, NTOK / 128), 256, 0, stream>>>(
        hln, wfc_b, bfc, nullptr, nullptr, fcb, NTOK, INTERN, HID, 1);

    // out = fcb @ wproj + bproj + x1 (fp32)
    gemm_bf16_kernel<<<dim3(HID / 128, NTOK / 128), 256, 0, stream>>>(
        fcb, wproj_b, bproj, x1, (float*)d_out, nullptr, NTOK, HID, INTERN, 0);
}
